// Block_89283780149784
// MI455X (gfx1250) — matrix operand order repaired and checked by reading, not compiled, not run on hardware
//
#include <hip/hip_runtime.h>
#include <math.h>

typedef __attribute__((ext_vector_type(16))) _Float16 v16h;
typedef __attribute__((ext_vector_type(8)))  _Float16 v8h;
typedef __attribute__((ext_vector_type(8)))  float    v8f;

#define T_   10
#define B_   128
#define C_   512
#define N_   16
#define H_   16
#define D_   32
#define HID_ 2048
#define P_   (T_ * B_ * N_)   // 20480 positions (t,b,n)
#define BN_  (B_ * N_)        // 2048

// ---------------------------------------------------------------------------
// Elementwise helpers
// ---------------------------------------------------------------------------

__global__ void f32_to_f16_kernel(const float* __restrict__ a,
                                  _Float16* __restrict__ b, int n) {
  int i = blockIdx.x * blockDim.x + threadIdx.x;
  if (i < n) b[i] = (_Float16)a[i];
}

// u = scale*conv + shift,  scale = g*rsqrt(v+eps),  shift = beta + scale*(bias - mean)
__global__ void bnprep_kernel(const float* __restrict__ bnp,
                              const float* __restrict__ bias,
                              float* __restrict__ scale,
                              float* __restrict__ shift, int Cc) {
  int o = blockIdx.x * blockDim.x + threadIdx.x;
  if (o >= Cc) return;
  float g = bnp[o], be = bnp[Cc + o], m = bnp[2 * Cc + o], v = bnp[3 * Cc + o];
  float sc = g * rsqrtf(v + 1e-5f);
  float bi = bias ? bias[o] : 0.0f;
  scale[o] = sc;
  shift[o] = be + sc * (bi - m);
}

// x (T,B,C,N) f32 natural -> Xp[p][c] f16 with p = (t*B+b)*N + n (c contiguous)
__global__ void pack_x_kernel(const float* __restrict__ x,
                              _Float16* __restrict__ Xp) {
  int idx = blockIdx.x * blockDim.x + threadIdx.x;
  if (idx >= T_ * B_ * C_ * N_) return;
  int n  = idx & 15;
  int c  = (idx >> 4) & (C_ - 1);
  int tb = idx >> 13;                       // / (C_*N_)
  Xp[(size_t)(tb * 16 + n) * C_ + c] = (_Float16)x[idx];
}

// x1h[p][c] = f16( x + Sssa )   (MLP input; residual stays implicit)
__global__ void resid_pack_kernel(const float* __restrict__ x,
                                  const _Float16* __restrict__ Sssa,
                                  _Float16* __restrict__ X1h) {
  int idx = blockIdx.x * blockDim.x + threadIdx.x;   // over P_*C_, idx = p*C_ + c
  if (idx >= P_ * C_) return;
  int c = idx & (C_ - 1);
  int p = idx >> 9;
  int n = p & 15, tb = p >> 4;
  float v = x[((size_t)tb * C_ + c) * 16 + n] + (float)Sssa[idx];
  X1h[idx] = (_Float16)v;
}

// out = x + Sssa + Smlp, written in natural (T,B,C,N) f32 layout
__global__ void final_kernel(const float* __restrict__ x,
                             const _Float16* __restrict__ Sssa,
                             const _Float16* __restrict__ Smlp,
                             float* __restrict__ out) {
  int idx = blockIdx.x * blockDim.x + threadIdx.x;   // over P_*C_
  if (idx >= P_ * C_) return;
  int c = idx & (C_ - 1);
  int p = idx >> 9;
  int n = p & 15, tb = p >> 4;
  size_t xi = ((size_t)tb * C_ + c) * 16 + n;
  out[xi] = x[xi] + (float)Sssa[idx] + (float)Smlp[idx];
}

// Multi-step LIF scan over T (tau=2, thr=1): reads preact f16 U[p][O],
// writes spikes f16 S[p][O]. One thread per (b,n,o).
__global__ void lif_scan_kernel(const _Float16* __restrict__ U,
                                _Float16* __restrict__ S, int O) {
  int idx = blockIdx.x * blockDim.x + threadIdx.x;   // over BN_*O
  int o  = idx % O;
  int bn = idx / O;
  float mem = 0.0f;
#pragma unroll
  for (int t = 0; t < T_; ++t) {
    size_t off = (size_t)(t * BN_ + bn) * O + o;
    float inp = (float)U[off];
    mem = 0.5f * (mem + inp);                 // mem + (inp-mem)/tau
    float s = mem > 1.0f ? 1.0f : 0.0f;       // heaviside(mem - thr)
    mem *= (1.0f - s);
    S[off] = (_Float16)s;
  }
}

// ---------------------------------------------------------------------------
// Register-blocked WMMA GEMM with fused BN epilogue:
//   U[p][o] = f16( scale[o] * sum_k W[o][k]*Act[p][k] + shift[o] )
// Each wave computes a 64(M=O) x 64(N=P) block as 4x4 WMMA tiles:
// per K-step it loads 4 A-fragments + 4 B-fragments and issues 16 WMMAs
// (1.0 b128 load per WMMA; operands reused 4x each in registers).
// 4 waves per block cover 256 rows of O.
// ---------------------------------------------------------------------------
__global__ __launch_bounds__(128) void gemm_bn_f16_kernel(
    const _Float16* __restrict__ W,    // [O][K] f16
    const _Float16* __restrict__ Act,  // [P][K] f16
    const float* __restrict__ scale, const float* __restrict__ shift,
    _Float16* __restrict__ U,          // [P][O] f16
    int O, int K) {
  const int lane = threadIdx.x & 31;
  const int wave = threadIdx.x >> 5;
  const int pB = blockIdx.x * 64;                  // P base (N dim)
  const int oB = (blockIdx.y * 4 + wave) * 64;     // O base (M dim) per wave
  const int m  = lane & 15;   // A-row / B-column within a 16-tile
  const int kh = lane >> 4;   // K-half select

  const _Float16* wbase = W   + (size_t)(oB + m) * K + kh * 16;
  const _Float16* abase = Act + (size_t)(pB + m) * K + kh * 16;

  v8f acc[4][4] = {};   // [mi][ni]
  for (int k0 = 0; k0 < K; k0 += 32) {
    v16h afrag[4], bfrag[4];
#pragma unroll
    for (int mi = 0; mi < 4; ++mi) {
      const _Float16* wr = wbase + (size_t)(mi * 16) * K + k0;
      v8h a0 = *(const v8h*)(wr);        // K = kh*16 + 0..7
      v8h a1 = *(const v8h*)(wr + 8);    // K = kh*16 + 8..15
      afrag[mi] = __builtin_shufflevector(a0, a1, 0, 1, 2, 3, 4, 5, 6, 7,
                                          8, 9, 10, 11, 12, 13, 14, 15);
    }
#pragma unroll
    for (int ni = 0; ni < 4; ++ni)
      bfrag[ni] = *(const v16h*)(abase + (size_t)(ni * 16) * K + k0);
#pragma unroll
    for (int mi = 0; mi < 4; ++mi)
#pragma unroll
      for (int ni = 0; ni < 4; ++ni)
        acc[mi][ni] = __builtin_amdgcn_wmma_f32_16x16x32_f16(
            false, afrag[mi], false, bfrag[ni], (short)0, acc[mi][ni],
            false, false);
  }

#pragma unroll
  for (int mi = 0; mi < 4; ++mi) {
    const int ob = oB + mi * 16 + kh * 8;          // rows M = kh*8 .. kh*8+7
    float sc[8], sh[8];
#pragma unroll
    for (int j = 0; j < 8; ++j) { sc[j] = scale[ob + j]; sh[j] = shift[ob + j]; }
#pragma unroll
    for (int ni = 0; ni < 4; ++ni) {
      const int p = pB + ni * 16 + m;
      _Float16* uptr = U + (size_t)p * O + ob;
#pragma unroll
      for (int j = 0; j < 8; ++j)
        uptr[j] = (_Float16)(acc[mi][ni][j] * sc[j] + sh[j]);
    }
  }
}

// ---------------------------------------------------------------------------
// Attention temporal scan. One wave per (b,h).
// q/k/v spikes live in [p][c] layout; reshape mapping (raw torch reshape):
//   q[t,b,h,n,d] -> c = 32n + 2h + (d>>4), ncol = d&15, p = (t*B+b)*16 + ncol
// Output Y is the pre-attn-LIF tensor in [p][c] layout with c = 32h + d.
// ---------------------------------------------------------------------------
__device__ __forceinline__ size_t spk_off(int t, int b, int h, int n, int d) {
  const int c = 32 * n + 2 * h + (d >> 4);
  const int ncol = d & 15;
  return (size_t)((t * B_ + b) * 16 + ncol) * C_ + c;
}

__global__ __launch_bounds__(32) void attn_scan_kernel(
    const _Float16* __restrict__ Sq, const _Float16* __restrict__ Sk,
    const _Float16* __restrict__ Sv, const float* __restrict__ tiw,
    const float* __restrict__ tib, _Float16* __restrict__ Y) {
  __shared__ float s2L[16][33];       // current query spikes (N x D)
  __shared__ float attnL[16][17];     // scaled attention (N x N)
  __shared__ float wL[16 * 16 * 5];   // ti conv weights [n][n'][tap]
  __shared__ float bL[16];

  const int l  = threadIdx.x;
  const int bh = blockIdx.x;
  const int b  = bh >> 4;
  const int h  = bh & 15;

  for (int i = l; i < 16 * 16 * 5; i += 32) wL[i] = tiw[i];
  if (l < 16) bL[l] = tib[l];

  const int en  = l & 15;             // element rows owned by this lane
  const int edb = (l >> 4) * 16;      // d-range base (0 or 16)

  float mem1[16], mem2[16];
#pragma unroll
  for (int i = 0; i < 16; ++i) { mem1[i] = 0.0f; mem2[i] = 0.0f; }

  // carry init: q_ti = q[0]
#pragma unroll
  for (int i = 0; i < 16; ++i)
    s2L[en][edb + i] = (float)Sq[spk_off(0, b, h, en, edb + i)];
  __syncthreads();

  const int m  = l & 15;
  const int kh = l >> 4;

  for (int t = 0; t < T_; ++t) {
    if (t > 0) {
      // 5-tap conv over D, mixing across N channels, on previous s2
      float cin[16];
#pragma unroll
      for (int i = 0; i < 16; ++i) {
        const int d = edb + i;
        float acc = bL[en];
        for (int n2 = 0; n2 < 16; ++n2) {
#pragma unroll
          for (int j = 0; j < 5; ++j) {
            const int dd = d + j - 2;
            if (dd >= 0 && dd < 32) acc += wL[en * 80 + n2 * 5 + j] * s2L[n2][dd];
          }
        }
        cin[i] = acc;
      }
      __syncthreads();   // finished reading old s2L
      float s2v[16];
#pragma unroll
      for (int i = 0; i < 16; ++i) {
        mem1[i] = 0.5f * (mem1[i] + cin[i]);
        const float s1 = mem1[i] > 1.0f ? 1.0f : 0.0f;
        mem1[i] *= (1.0f - s1);
        const float qt = (float)Sq[spk_off(t, b, h, en, edb + i)];
        const float mix = 0.5f * s1 + 0.5f * qt;   // alpha_mix = 0.5
        mem2[i] = 0.5f * (mem2[i] + mix);
        const float s2 = mem2[i] > 1.0f ? 1.0f : 0.0f;
        mem2[i] *= (1.0f - s2);
        s2v[i] = s2;
      }
#pragma unroll
      for (int i = 0; i < 16; ++i) s2L[en][edb + i] = s2v[i];
      __syncthreads();
    }

    // attn = (s2 @ kt^T) * 0.25   (M=N=16, K=D=32)
    v16h a;
#pragma unroll
    for (int j = 0; j < 8; ++j) {
      a[j]     = (_Float16)s2L[m][kh * 8 + j];
      a[8 + j] = (_Float16)s2L[m][16 + kh * 8 + j];
    }
    v16h bk;
#pragma unroll
    for (int i = 0; i < 16; ++i)
      bk[i] = Sk[spk_off(t, b, h, m, kh * 8 + (i & 7) + (i >> 3) * 16)];   // B[k=d][col=n2]
    v8f c0 = {};
    c0 = __builtin_amdgcn_wmma_f32_16x16x32_f16(false, a, false, bk,
                                                (short)0, c0, false, false);
    __syncthreads();
#pragma unroll
    for (int j = 0; j < 8; ++j) attnL[kh * 8 + j][m] = c0[j] * 0.25f;
    __syncthreads();

    // out = attn @ vt  (K=16 zero-padded to 32; two 16-wide d halves)
    v16h a2;
#pragma unroll
    for (int j = 0; j < 8; ++j) {
      a2[j]     = (_Float16)attnL[m][kh * 8 + j];
      a2[8 + j] = (_Float16)0.0f;   // K >= 16 pad
    }
#pragma unroll
    for (int dh = 0; dh < 2; ++dh) {
      v16h bv;
#pragma unroll
      for (int i = 0; i < 16; ++i)
        bv[i] = (i < 8) ? Sv[spk_off(t, b, h, kh * 8 + i, dh * 16 + m)]
                        : (_Float16)0.0f;  // K >= 16 pad
      v8f co = {};
      co = __builtin_amdgcn_wmma_f32_16x16x32_f16(false, a2, false, bv,
                                                  (short)0, co, false, false);
#pragma unroll
      for (int j = 0; j < 8; ++j) {
        const int n = kh * 8 + j;
        const int d = dh * 16 + m;
        Y[(size_t)((t * B_ + b) * 16 + n) * C_ + 32 * h + d] = (_Float16)co[j];
      }
    }
    __syncthreads();
  }
}

// ---------------------------------------------------------------------------
// Launch
// ---------------------------------------------------------------------------
extern "C" void kernel_launch(void* const* d_in, const int* in_sizes, int n_in,
                              void* d_out, int out_size, void* d_ws, size_t ws_size,
                              hipStream_t stream) {
  (void)in_sizes; (void)n_in; (void)out_size; (void)ws_size;
  const float* x     = (const float*)d_in[0];
  const float* Wq    = (const float*)d_in[1];
  const float* Wk    = (const float*)d_in[2];
  const float* Wv    = (const float*)d_in[3];
  const float* Wproj = (const float*)d_in[4];
  const float* bn_q  = (const float*)d_in[5];
  const float* bn_k  = (const float*)d_in[6];
  const float* bn_v  = (const float*)d_in[7];
  const float* bn_p  = (const float*)d_in[8];
  const float* ti_w  = (const float*)d_in[9];
  const float* ti_b  = (const float*)d_in[10];
  const float* W1    = (const float*)d_in[11];
  const float* b1    = (const float*)d_in[12];
  const float* bn1   = (const float*)d_in[13];
  const float* W2    = (const float*)d_in[14];
  const float* b2    = (const float*)d_in[15];
  const float* bn2   = (const float*)d_in[16];
  float* out = (float*)d_out;

  char* wp = (char*)d_ws;
  auto carve = [&](size_t bytes) -> char* {
    char* r = wp;
    wp += (bytes + 255) & ~(size_t)255;
    return r;
  };

  const size_t PC  = (size_t)P_ * C_;
  const size_t PH  = (size_t)P_ * HID_;

  _Float16* Xp    = (_Float16*)carve(PC * 2);
  _Float16* U     = (_Float16*)carve(PC * 2);    // reused preact buffer
  _Float16* Sq    = (_Float16*)carve(PC * 2);
  _Float16* Sk    = (_Float16*)carve(PC * 2);
  _Float16* Sv    = (_Float16*)carve(PC * 2);
  _Float16* Sattn = (_Float16*)carve(PC * 2);
  _Float16* Sssa  = (_Float16*)carve(PC * 2);
  _Float16* X1h   = (_Float16*)carve(PC * 2);
  _Float16* Smlp  = (_Float16*)carve(PC * 2);
  _Float16* Uh    = (_Float16*)carve(PH * 2);
  _Float16* Sh    = (_Float16*)carve(PH * 2);
  _Float16* Wq16  = (_Float16*)carve((size_t)C_ * C_ * 2);
  _Float16* Wk16  = (_Float16*)carve((size_t)C_ * C_ * 2);
  _Float16* Wv16  = (_Float16*)carve((size_t)C_ * C_ * 2);
  _Float16* Wp16  = (_Float16*)carve((size_t)C_ * C_ * 2);
  _Float16* W116  = (_Float16*)carve((size_t)HID_ * C_ * 2);
  _Float16* W216  = (_Float16*)carve((size_t)C_ * HID_ * 2);
  float* scq = (float*)carve(C_ * 4);   float* shq = (float*)carve(C_ * 4);
  float* sck = (float*)carve(C_ * 4);   float* shk = (float*)carve(C_ * 4);
  float* scv = (float*)carve(C_ * 4);   float* shv = (float*)carve(C_ * 4);
  float* scp = (float*)carve(C_ * 4);   float* shp = (float*)carve(C_ * 4);
  float* sc1 = (float*)carve(HID_ * 4); float* sh1 = (float*)carve(HID_ * 4);
  float* sc2 = (float*)carve(C_ * 4);   float* sh2 = (float*)carve(C_ * 4);

  // BN scale/shift precompute (fold conv bias for MLP layers)
  bnprep_kernel<<<(C_ + 255) / 256, 256, 0, stream>>>(bn_q, nullptr, scq, shq, C_);
  bnprep_kernel<<<(C_ + 255) / 256, 256, 0, stream>>>(bn_k, nullptr, sck, shk, C_);
  bnprep_kernel<<<(C_ + 255) / 256, 256, 0, stream>>>(bn_v, nullptr, scv, shv, C_);
  bnprep_kernel<<<(C_ + 255) / 256, 256, 0, stream>>>(bn_p, nullptr, scp, shp, C_);
  bnprep_kernel<<<(HID_ + 255) / 256, 256, 0, stream>>>(bn1, b1, sc1, sh1, HID_);
  bnprep_kernel<<<(C_ + 255) / 256, 256, 0, stream>>>(bn2, b2, sc2, sh2, C_);

  // Weight conversion f32 -> f16
  f32_to_f16_kernel<<<(C_ * C_ + 255) / 256, 256, 0, stream>>>(Wq, Wq16, C_ * C_);
  f32_to_f16_kernel<<<(C_ * C_ + 255) / 256, 256, 0, stream>>>(Wk, Wk16, C_ * C_);
  f32_to_f16_kernel<<<(C_ * C_ + 255) / 256, 256, 0, stream>>>(Wv, Wv16, C_ * C_);
  f32_to_f16_kernel<<<(C_ * C_ + 255) / 256, 256, 0, stream>>>(Wproj, Wp16, C_ * C_);
  f32_to_f16_kernel<<<(HID_ * C_ + 255) / 256, 256, 0, stream>>>(W1, W116, HID_ * C_);
  f32_to_f16_kernel<<<(C_ * HID_ + 255) / 256, 256, 0, stream>>>(W2, W216, C_ * HID_);

  // Activations -> [p][c] f16
  pack_x_kernel<<<(int)(PC / 256), 256, 0, stream>>>(x, Xp);

  const dim3 gemmGridC(P_ / 64, C_ / 256);    // (320, 2), 4 waves * 64 rows
  const dim3 gemmGridH(P_ / 64, HID_ / 256);  // (320, 8)
  const int lifBlocksC = (BN_ * C_) / 256;
  const int lifBlocksH = (BN_ * HID_) / 256;

  // q / k / v : GEMM + BN -> LIF spikes
  gemm_bn_f16_kernel<<<gemmGridC, 128, 0, stream>>>(Wq16, Xp, scq, shq, U, C_, C_);
  lif_scan_kernel<<<lifBlocksC, 256, 0, stream>>>(U, Sq, C_);
  gemm_bn_f16_kernel<<<gemmGridC, 128, 0, stream>>>(Wk16, Xp, sck, shk, U, C_, C_);
  lif_scan_kernel<<<lifBlocksC, 256, 0, stream>>>(U, Sk, C_);
  gemm_bn_f16_kernel<<<gemmGridC, 128, 0, stream>>>(Wv16, Xp, scv, shv, U, C_, C_);
  lif_scan_kernel<<<lifBlocksC, 256, 0, stream>>>(U, Sv, C_);

  // temporal attention scan -> Y (reuses U), then attn-LIF
  attn_scan_kernel<<<B_ * H_, 32, 0, stream>>>(Sq, Sk, Sv, ti_w, ti_b, U);
  lif_scan_kernel<<<lifBlocksC, 256, 0, stream>>>(U, Sattn, C_);

  // proj GEMM + BN -> LIF -> ssa spikes
  gemm_bn_f16_kernel<<<gemmGridC, 128, 0, stream>>>(Wp16, Sattn, scp, shp, U, C_, C_);
  lif_scan_kernel<<<lifBlocksC, 256, 0, stream>>>(U, Sssa, C_);

  // x1 = x + ssa (f16 pack for MLP input)
  resid_pack_kernel<<<(int)(PC / 256), 256, 0, stream>>>(x, Sssa, X1h);

  // MLP
  gemm_bn_f16_kernel<<<gemmGridH, 128, 0, stream>>>(W116, X1h, sc1, sh1, Uh, HID_, C_);
  lif_scan_kernel<<<lifBlocksH, 256, 0, stream>>>(Uh, Sh, HID_);
  gemm_bn_f16_kernel<<<gemmGridC, 128, 0, stream>>>(W216, Sh, sc2, sh2, U, C_, HID_);
  lif_scan_kernel<<<lifBlocksC, 256, 0, stream>>>(U, Smlp, C_);

  // out = x + ssa + mlp  in natural (T,B,C,N) layout
  final_kernel<<<(int)(PC / 256), 256, 0, stream>>>(x, Sssa, Smlp, out);
}
